// GCN_83640193122934
// MI455X (gfx1250) — compile-verified, matrix operand order checked by reading, not run on hardware
//
#include <hip/hip_runtime.h>
#include <hip/hip_bf16.h>
#include <stdint.h>

// ---------------- problem constants (match reference) ----------------
#define N_NODES 100000
#define N_EDGES 1600000
#define F_IN    256
#define HID     64
#define RT      5          // row tiles (of 16 rows) per block; 6250/5 = 1250 blocks

typedef __attribute__((ext_vector_type(16))) _Float16 v16h;
typedef __attribute__((ext_vector_type(8)))  float    v8f;
typedef __attribute__((ext_vector_type(4)))  float    v4f;   // clang vector: valid for nontemporal builtins

static __device__ __forceinline__ void atomicAddF32(float* p, float v) {
    __hip_atomic_fetch_add(p, v, __ATOMIC_RELAXED, __HIP_MEMORY_SCOPE_AGENT);
}

// ---------------- kernel 1: zero integer degree array ----------------
__global__ void k_deg_zero(int* __restrict__ deg) {
    int i = blockIdx.x * blockDim.x + threadIdx.x;
    if (i < N_NODES) deg[i] = 0;
}

// ---------------- kernel 2: in-degree count on dst -------------------
__global__ void k_deg_count(const int* __restrict__ ei, int* __restrict__ deg) {
    int e = blockIdx.x * blockDim.x + threadIdx.x;
    if (e < N_EDGES) {
        int d = ei[N_EDGES + e];  // dst row of edge_index
        atomicAdd(&deg[d], 1);
    }
}

// ---------------- kernel 3: dinv = rsqrt(deg + 1) --------------------
__global__ void k_dinv(const int* __restrict__ deg, float* __restrict__ dinv) {
    int i = blockIdx.x * blockDim.x + threadIdx.x;
    if (i < N_NODES) dinv[i] = __frsqrt_rn((float)deg[i] + 1.0f);
}

// ---------------- kernel 4: GEMM1  xw1 = x @ W1  (WMMA f16->f32) -----
// 128 threads = 4 waves; wave w owns column tile [16w, 16w+16).
// B fragments (all of W1 for this column tile) are gathered ONCE into
// registers (8 x v16h) and reused across RT row tiles -> no LDS in the
// inner loop, one v_wmma per K-tile.
__global__ __launch_bounds__(128) void k_gemm1_wmma(
        const float* __restrict__ x,
        const float* __restrict__ W1,
        float* __restrict__ xw1) {
    const int wave = threadIdx.x >> 5;   // 0..3 -> column tile
    const int lane = threadIdx.x & 31;
    const int half = lane >> 4;          // K-half selector per ISA 16-bit layout
    const int r    = lane & 15;          // A row / B column within tile
    const int n0   = wave * 16;

    // ---- gather B fragments into registers (coalesced, L2-hot W1) ----
    v16h bfrag[8];
    #pragma unroll
    for (int kt = 0; kt < 8; ++kt) {
        const int kbase = kt * 32;
        #pragma unroll
        for (int j = 0; j < 8; ++j)
            bfrag[kt][j]     = (_Float16)W1[(kbase + 8 * half + j) * HID + (n0 + r)];
        #pragma unroll
        for (int j = 0; j < 8; ++j)
            bfrag[kt][8 + j] = (_Float16)W1[(kbase + 16 + 8 * half + j) * HID + (n0 + r)];
    }

    for (int rt = 0; rt < RT; ++rt) {
        const int row0 = (blockIdx.x * RT + rt) * 16;
        const float* xrow = x + (size_t)(row0 + r) * F_IN;

        // prefetch next row tile's A row (streams x through L2 ahead of use)
        if (rt + 1 < RT) {
            const float* nxt = xrow + 16 * F_IN;
            __builtin_prefetch(nxt +   0, 0, 0);
            __builtin_prefetch(nxt +  64, 0, 0);
            __builtin_prefetch(nxt + 128, 0, 0);
            __builtin_prefetch(nxt + 192, 0, 0);
        }

        v8f acc = {0.f, 0.f, 0.f, 0.f, 0.f, 0.f, 0.f, 0.f};
        #pragma unroll
        for (int kt = 0; kt < 8; ++kt) {
            const float* seg0 = xrow + kt * 32 + 8 * half;       // K in [8h, 8h+8)
            const float* seg1 = seg0 + 16;                       // K in [16+8h, 16+8h+8)
            v4f a0 = __builtin_nontemporal_load((const v4f*)seg0);
            v4f a1 = __builtin_nontemporal_load((const v4f*)(seg0 + 4));
            v4f a2 = __builtin_nontemporal_load((const v4f*)seg1);
            v4f a3 = __builtin_nontemporal_load((const v4f*)(seg1 + 4));
            v16h a;
            a[0]  = (_Float16)a0.x; a[1]  = (_Float16)a0.y;
            a[2]  = (_Float16)a0.z; a[3]  = (_Float16)a0.w;
            a[4]  = (_Float16)a1.x; a[5]  = (_Float16)a1.y;
            a[6]  = (_Float16)a1.z; a[7]  = (_Float16)a1.w;
            a[8]  = (_Float16)a2.x; a[9]  = (_Float16)a2.y;
            a[10] = (_Float16)a2.z; a[11] = (_Float16)a2.w;
            a[12] = (_Float16)a3.x; a[13] = (_Float16)a3.y;
            a[14] = (_Float16)a3.z; a[15] = (_Float16)a3.w;

            acc = __builtin_amdgcn_wmma_f32_16x16x32_f16(
                /*neg_a=*/false, a, /*neg_b=*/false, bfrag[kt],
                /*c_mod=*/(short)0, acc, /*reuse_a=*/false, /*reuse_b=*/false);
        }

        // D layout: VGPR i -> M = i + 8*half, N = lane&15
        #pragma unroll
        for (int i = 0; i < 8; ++i) {
            int m = i + 8 * half;
            xw1[(size_t)(row0 + m) * HID + (n0 + r)] = acc[i];
        }
    }
}

// ---------------- kernel 5: agg1 init (self-loop + bias) -------------
__global__ void k_agg1_init(const float* __restrict__ xw1,
                            const float* __restrict__ dinv,
                            const float* __restrict__ b1,
                            float* __restrict__ agg1) {
    size_t t = (size_t)blockIdx.x * blockDim.x + threadIdx.x;
    if (t < (size_t)N_NODES * HID) {
        int node = (int)(t >> 6);
        int f    = (int)(t & 63);
        float di = dinv[node];
        agg1[t] = xw1[t] * (di * di) + b1[f];
    }
}

// ---------------- kernel 6: layer-1 message scatter ------------------
// one wave per edge; lane handles 2 contiguous features (float2).
// xw1 + agg1 (25.6MB each) are L2-resident on MI455X (192MB L2), so the
// random gather + atomic scatter run at L2 bandwidth.
__global__ void k_agg1_edges(const int* __restrict__ ei,
                             const float* __restrict__ dinv,
                             const float* __restrict__ xw1,
                             float* __restrict__ agg1) {
    size_t t = (size_t)blockIdx.x * blockDim.x + threadIdx.x;
    int e    = (int)(t >> 5);
    int lane = (int)(t & 31);
    if (e >= N_EDGES) return;
    // e is wave-uniform: force scalar so index/dinv fetches use the SMEM path
    e = __builtin_amdgcn_readfirstlane(e);
    int s = ei[e];
    int d = ei[N_EDGES + e];
    float norm = dinv[s] * dinv[d];
    const float2 v = *(const float2*)(xw1 + (size_t)s * HID + lane * 2);
    float* out = agg1 + (size_t)d * HID + lane * 2;
    atomicAddF32(out + 0, v.x * norm);
    atomicAddF32(out + 1, v.y * norm);
}

// ---------------- kernel 7: relu + GEMM2 + agg2 init -----------------
__global__ void k_relu_gemm2(const float* __restrict__ agg1,
                             const float* __restrict__ W2,
                             const float* __restrict__ b2,
                             const float* __restrict__ dinv,
                             float* __restrict__ hw2,
                             float* __restrict__ agg2) {
    int i = blockIdx.x * blockDim.x + threadIdx.x;
    if (i >= N_NODES) return;
    const float* row = agg1 + (size_t)i * HID;
    float o0 = 0.f, o1 = 0.f;
    #pragma unroll 8
    for (int k = 0; k < HID; ++k) {
        float h = fmaxf(row[k], 0.f);  // ReLU (dropout is identity in eval)
        o0 += h * W2[k * 2 + 0];
        o1 += h * W2[k * 2 + 1];
    }
    hw2[(size_t)i * 2 + 0] = o0;
    hw2[(size_t)i * 2 + 1] = o1;
    float di = dinv[i];
    float sl = di * di;
    agg2[(size_t)i * 2 + 0] = o0 * sl + b2[0];
    agg2[(size_t)i * 2 + 1] = o1 * sl + b2[1];
}

// ---------------- kernel 8: layer-2 message scatter ------------------
__global__ void k_agg2_edges(const int* __restrict__ ei,
                             const float* __restrict__ dinv,
                             const float* __restrict__ hw2,
                             float* __restrict__ agg2) {
    int e = blockIdx.x * blockDim.x + threadIdx.x;
    if (e >= N_EDGES) return;
    int s = ei[e];
    int d = ei[N_EDGES + e];
    float norm = dinv[s] * dinv[d];
    const float2 v = *(const float2*)(hw2 + (size_t)s * 2);
    float* out = agg2 + (size_t)d * 2;
    atomicAddF32(out + 0, v.x * norm);
    atomicAddF32(out + 1, v.y * norm);
}

// ---------------- kernel 9: 2-class softmax --------------------------
__global__ void k_softmax2(const float* __restrict__ agg2,
                           float* __restrict__ out) {
    int i = blockIdx.x * blockDim.x + threadIdx.x;
    if (i >= N_NODES) return;
    float a0 = agg2[(size_t)i * 2 + 0];
    float a1 = agg2[(size_t)i * 2 + 1];
    float m  = fmaxf(a0, a1);
    float e0 = __expf(a0 - m);
    float e1 = __expf(a1 - m);
    float inv = 1.0f / (e0 + e1);
    out[(size_t)i * 2 + 0] = e0 * inv;
    out[(size_t)i * 2 + 1] = e1 * inv;
}

// ---------------- launcher -------------------------------------------
extern "C" void kernel_launch(void* const* d_in, const int* in_sizes, int n_in,
                              void* d_out, int out_size, void* d_ws, size_t ws_size,
                              hipStream_t stream) {
    const float* x  = (const float*)d_in[0];
    const int*   ei = (const int*)d_in[1];      // [2, E] src row then dst row
    const float* W1 = (const float*)d_in[2];
    const float* b1 = (const float*)d_in[3];
    const float* W2 = (const float*)d_in[4];
    const float* b2 = (const float*)d_in[5];
    float* out = (float*)d_out;

    // workspace layout (bytes, 4KB-aligned slabs)
    char* ws = (char*)d_ws;
    size_t o = 0;
    auto alloc = [&](size_t bytes) {
        char* p = ws + o;
        o += (bytes + 4095) & ~(size_t)4095;
        return p;
    };
    int*   deg  = (int*)  alloc((size_t)N_NODES * sizeof(int));
    float* dinv = (float*)alloc((size_t)N_NODES * sizeof(float));
    float* xw1  = (float*)alloc((size_t)N_NODES * HID * sizeof(float));
    float* agg1 = (float*)alloc((size_t)N_NODES * HID * sizeof(float));
    float* hw2  = (float*)alloc((size_t)N_NODES * 2 * sizeof(float));
    float* agg2 = (float*)alloc((size_t)N_NODES * 2 * sizeof(float));
    (void)ws_size; (void)in_sizes; (void)n_in; (void)out_size;

    const int B = 256;
    dim3 blk(B);

    // degrees + dinv
    k_deg_zero <<<(N_NODES + B - 1) / B, blk, 0, stream>>>(deg);
    k_deg_count<<<(N_EDGES + B - 1) / B, blk, 0, stream>>>(ei, deg);
    k_dinv     <<<(N_NODES + B - 1) / B, blk, 0, stream>>>(deg, dinv);

    // layer 1 GEMM via WMMA: 6250 row tiles / RT per block
    k_gemm1_wmma<<<(N_NODES / 16) / RT, dim3(128), 0, stream>>>(x, W1, xw1);

    // layer 1 aggregation
    size_t t1 = (size_t)N_NODES * HID;
    k_agg1_init<<<(unsigned)((t1 + B - 1) / B), blk, 0, stream>>>(xw1, dinv, b1, agg1);
    size_t t2 = (size_t)N_EDGES * 32;  // one wave per edge
    k_agg1_edges<<<(unsigned)((t2 + B - 1) / B), blk, 0, stream>>>(ei, dinv, xw1, agg1);

    // relu + layer 2 GEMM + self-loop init
    k_relu_gemm2<<<(N_NODES + B - 1) / B, blk, 0, stream>>>(agg1, W2, b2, dinv, hw2, agg2);

    // layer 2 aggregation + softmax
    k_agg2_edges<<<(N_EDGES + B - 1) / B, blk, 0, stream>>>(ei, dinv, hw2, agg2);
    k_softmax2  <<<(N_NODES + B - 1) / B, blk, 0, stream>>>(agg2, out);
}
